// webformerAttention_38268158607975
// MI455X (gfx1250) — matrix operand order repaired and checked by reading, not compiled, not run on hardware
//
#include <hip/hip_runtime.h>

// ---------------------------------------------------------------------------
// Types / helpers
// ---------------------------------------------------------------------------
typedef __attribute__((ext_vector_type(16))) __bf16 v16bf;
typedef __attribute__((ext_vector_type(8)))  float  v8f;

typedef unsigned short u16;
typedef unsigned int   u32;

__device__ __forceinline__ u16 f2bf(float f) {
  u32 u = __float_as_uint(f);
  u32 r = u + 0x7FFFu + ((u >> 16) & 1u);   // round-to-nearest-even
  return (u16)(r >> 16);
}
__device__ __forceinline__ float bf2f(u16 h) {
  return __uint_as_float(((u32)h) << 16);
}

union Frag {
  uint4 q[2];
  v16bf v;
};

// ---------------------------------------------------------------------------
// NT bf16 GEMM: C[m,n] = sum_k A[m,k] * B[n,k]
// A: (M,K) bf16 row-major, B: (N,K) bf16 row-major, C: f32 or bf16.
// Block 128 threads = 4 waves, block tile 64x64, wave tile 32x32, K-step 32.
// flags: bit0 = accumulate into f32 C, bit1 = write bf16 C (no accumulate)
// ---------------------------------------------------------------------------
#define BM 64
#define BN 64
#define BK 32

__global__ __launch_bounds__(128)
void gemm_nt_bf16(const u16* __restrict__ A, long ldA, long bsA,
                  const u16* __restrict__ B, long ldB, long bsB,
                  void* __restrict__ Cv, long ldC, long bsC,
                  int M, int N, int K, int flags)
{
  __shared__ u16 sA[BM * BK];   // 4 KB
  __shared__ u16 sB[BN * BK];   // 4 KB

  const int tid  = threadIdx.x;
  const int lane = tid & 31;
  const int wave = tid >> 5;       // 0..3
  const int wm   = wave & 1;       // wave row (2)
  const int wn   = wave >> 1;      // wave col (2)

  const long rowBase = (long)blockIdx.y * BM;
  const long colBase = (long)blockIdx.x * BN;

  A += bsA * blockIdx.z;
  B += bsB * blockIdx.z;
  const long cb = bsC * blockIdx.z;

  // global->LDS staging: each thread moves 32B of A and 32B of B per k-step
  const int ldRow = tid >> 1;            // 0..63
  const int ldCol = (tid & 1) * 16;      // 0 or 16

  v8f cacc[2][2];
  #pragma unroll
  for (int a = 0; a < 2; ++a)
    #pragma unroll
    for (int b = 0; b < 2; ++b)
      cacc[a][b] = (v8f){0.f,0.f,0.f,0.f,0.f,0.f,0.f,0.f};

  const int h = lane >> 4;   // lane half
  const int r = lane & 15;

  for (int k0 = 0; k0 < K; k0 += BK) {
    const u16* ga = A + (rowBase + ldRow) * ldA + k0 + ldCol;
    const u16* gb = B + (colBase + ldRow) * ldB + k0 + ldCol;
    uint4 a0 = *(const uint4*)ga;
    uint4 a1 = *(const uint4*)(ga + 8);
    uint4 b0 = *(const uint4*)gb;
    uint4 b1 = *(const uint4*)(gb + 8);
    __syncthreads();
    *(uint4*)&sA[ldRow * BK + ldCol]     = a0;
    *(uint4*)&sA[ldRow * BK + ldCol + 8] = a1;
    *(uint4*)&sB[ldRow * BK + ldCol]     = b0;
    *(uint4*)&sB[ldRow * BK + ldCol + 8] = b1;
    __syncthreads();

    Frag af[2], bf[2];
    #pragma unroll
    for (int a = 0; a < 2; ++a) {
      int row = wm * 32 + a * 16 + r;
      // A-frag: chunks K[h*8 .. h*8+7] and K[16+h*8 .. 16+h*8+7]
      af[a].q[0] = *(const uint4*)&sA[row * BK + h * 8];
      af[a].q[1] = *(const uint4*)&sA[row * BK + 16 + h * 8];
    }
    #pragma unroll
    for (int b = 0; b < 2; ++b) {
      int col = wn * 32 + b * 16 + r;
      int kb  = h * 8;       // B-frag: chunks K[h*8 .. h*8+7] and K[16+h*8 .. 16+h*8+7]
      bf[b].q[0] = *(const uint4*)&sB[col * BK + kb];
      bf[b].q[1] = *(const uint4*)&sB[col * BK + kb + 16];
    }

    cacc[0][0] = __builtin_amdgcn_wmma_f32_16x16x32_bf16(false, af[0].v, false, bf[0].v, (short)0, cacc[0][0], false, false);
    cacc[0][1] = __builtin_amdgcn_wmma_f32_16x16x32_bf16(false, af[0].v, false, bf[1].v, (short)0, cacc[0][1], false, false);
    cacc[1][0] = __builtin_amdgcn_wmma_f32_16x16x32_bf16(false, af[1].v, false, bf[0].v, (short)0, cacc[1][0], false, false);
    cacc[1][1] = __builtin_amdgcn_wmma_f32_16x16x32_bf16(false, af[1].v, false, bf[1].v, (short)0, cacc[1][1], false, false);
  }

  const bool acc = (flags & 1) != 0;
  const bool obf = (flags & 2) != 0;

  #pragma unroll
  for (int a = 0; a < 2; ++a) {
    #pragma unroll
    for (int b = 0; b < 2; ++b) {
      long row0 = rowBase + wm * 32 + a * 16 + h * 8;
      long col  = colBase + wn * 32 + b * 16 + r;
      #pragma unroll
      for (int v = 0; v < 8; ++v) {
        long idx = cb + (row0 + v) * ldC + col;
        float val = cacc[a][b][v];
        if (obf) {
          ((u16*)Cv)[idx] = f2bf(val);
        } else {
          float* Cp = (float*)Cv;
          if (acc) Cp[idx] += val; else Cp[idx] = val;
        }
      }
    }
  }
}

// ---------------------------------------------------------------------------
// Elementwise / small kernels
// ---------------------------------------------------------------------------
__global__ void cvt_f32_bf16(const float* __restrict__ in, u16* __restrict__ out, long n) {
  long i = (long)blockIdx.x * blockDim.x + threadIdx.x;
  if (i < n) out[i] = f2bf(in[i]);
}

// in: (M,N) f32 row-major -> out: (N,M) bf16 row-major
__global__ void transpose_f32_bf16(const float* __restrict__ in, u16* __restrict__ out,
                                   int M, int N) {
  long i = (long)blockIdx.x * blockDim.x + threadIdx.x;
  if (i >= (long)M * N) return;
  int m = (int)(i / N), n = (int)(i % N);
  out[(long)n * M + m] = f2bf(in[i]);
}

// batched bf16 transpose: in (M,N) -> out (N,M), grid.z = batch
__global__ void transpose_bf16(const u16* __restrict__ in, u16* __restrict__ out,
                               int M, int N, long bsIn, long bsOut) {
  long i = (long)blockIdx.x * blockDim.x + threadIdx.x;
  if (i >= (long)M * N) return;
  int m = (int)(i / N), n = (int)(i % N);
  out[(long)blockIdx.z * bsOut + (long)n * M + m] = in[(long)blockIdx.z * bsIn + i];
}

// qp[row, p] = dot(t2t_q[row,:], posemb[p,:]), row in [0, B*Lt), p in [0,13)
__global__ void qp_kernel(const u16* __restrict__ q, const float* __restrict__ pos,
                          float* __restrict__ qp, long total) {
  long i = (long)blockIdx.x * blockDim.x + threadIdx.x;
  if (i >= total) return;
  int  p   = (int)(i % 13);
  long row = i / 13;
  const u16*   qr = q + row * 384;
  const float* pr = pos + (long)p * 384;
  float s = 0.f;
  for (int d = 0; d < 384; ++d) s += bf2f(qr[d]) * pr[d];
  qp[i] = s;
}

// Fused bias/mask + softmax; one block per row. Writes P as bf16.
// mode 0: plain (T2H)            e = s*scale
// mode 1: float mask (H2T)       e = s*scale - 1e4*fmask
// mode 2: H2H                    e = (s + bias[imask])*scale - 1e4*(imask==0)
// mode 3: T2T                    e = (s + bias[clip(j-i)+6])*scale - 1e4*fmask
__global__ __launch_bounds__(256)
void softmax_kernel(const float* __restrict__ S, u16* __restrict__ P,
                    const int* __restrict__ imask, const float* __restrict__ fmask,
                    const float* __restrict__ bias, int biasW,
                    int Lq, int Lk, int mode)
{
  const int b = blockIdx.y;
  const int i = blockIdx.x;
  const long rowOff = ((long)b * Lq + i) * (long)Lk;
  const float scale = 0.051031036307982884f;   // 1/sqrt(384)
  const float* biasRow = bias ? bias + ((long)b * Lq + i) * biasW : nullptr;

  const int tid  = threadIdx.x;
  const int nPer = Lk >> 8;                    // Lk is 512 or 1024 here
  float e[4];
  float mx = -3.0e38f;

  for (int t = 0; t < nPer; ++t) {
    int j = tid + t * 256;
    float s = S[rowOff + j];
    if (mode == 2) {
      int idx = imask[rowOff + j];
      s = (s + biasRow[idx]) * scale;
      if (idx == 0) s -= 10000.f;
    } else if (mode == 3) {
      int d = j - i; d = d < -6 ? -6 : (d > 6 ? 6 : d);
      s = (s + biasRow[d + 6]) * scale - 10000.f * fmask[rowOff + j];
    } else if (mode == 1) {
      s = s * scale - 10000.f * fmask[rowOff + j];
    } else {
      s = s * scale;
    }
    e[t] = s;
    mx = fmaxf(mx, s);
  }

  __shared__ float red[256];
  red[tid] = mx; __syncthreads();
  for (int s = 128; s > 0; s >>= 1) {
    if (tid < s) red[tid] = fmaxf(red[tid], red[tid + s]);
    __syncthreads();
  }
  mx = red[0]; __syncthreads();

  float sum = 0.f;
  for (int t = 0; t < nPer; ++t) { e[t] = __expf(e[t] - mx); sum += e[t]; }
  red[tid] = sum; __syncthreads();
  for (int s = 128; s > 0; s >>= 1) {
    if (tid < s) red[tid] += red[tid + s];
    __syncthreads();
  }
  const float inv = 1.0f / red[0];

  for (int t = 0; t < nPer; ++t) {
    int j = tid + t * 256;
    P[rowOff + j] = f2bf(e[t] * inv);
  }
}

// ---------------------------------------------------------------------------
// Launch
// ---------------------------------------------------------------------------
extern "C" void kernel_launch(void* const* d_in, const int* in_sizes, int n_in,
                              void* d_out, int out_size, void* d_ws, size_t ws_size,
                              hipStream_t stream)
{
  (void)in_sizes; (void)n_in; (void)out_size; (void)ws_size;
  constexpr int  Bb = 4, Lt = 1024, Lh = 512, Dd = 384, Ee = 64, Pw = 13;
  constexpr long Mt = (long)Bb * Lt, Mh = (long)Bb * Lh;

  const float* text = (const float*)d_in[0];
  const float* html = (const float*)d_in[1];
  const int*   H2H  = (const int*)d_in[2];
  const float* T2T  = (const float*)d_in[3];
  const float* H2T  = (const float*)d_in[4];
  // d_in[5] (T2Hmask) is never applied by the reference.
  const float* pose = (const float*)d_in[6];
  const float* edge = (const float*)d_in[7];
  const float* W[10];
  for (int i = 0; i < 10; ++i) W[i] = (const float*)d_in[8 + i];
  float* out = (float*)d_out;

  // workspace carve-up
  char* base = (char*)d_ws; size_t off = 0;
  auto alloc = [&](size_t bytes) -> void* {
    void* p = base + off;
    off = (off + bytes + 255) & ~(size_t)255;
    return p;
  };
  u16* textb = (u16*)alloc((size_t)Mt * Dd * 2);
  u16* htmlb = (u16*)alloc((size_t)Mh * Dd * 2);
  u16* Wt    = (u16*)alloc((size_t)10 * Dd * Dd * 2);
  u16* edgeb = (u16*)alloc((size_t)Ee * Dd * 2);
  u16* Pt[5]; for (int i = 0; i < 5; ++i) Pt[i] = (u16*)alloc((size_t)Mt * Dd * 2);
  u16* Ph[5]; for (int i = 0; i < 5; ++i) Ph[i] = (u16*)alloc((size_t)Mh * Dd * 2);
  u16* VtT  = (u16*)alloc((size_t)Mt * Dd * 2);   // per-batch (Dd, Lt)
  u16* VhT  = (u16*)alloc((size_t)Mh * Dd * 2);   // per-batch (Dd, Lh)
  float* qe = (float*)alloc((size_t)Bb * Lh * Ee * 4);
  float* qp = (float*)alloc((size_t)Bb * Lt * Pw * 4);
  float* S  = (float*)alloc((size_t)Bb * Lt * Lt * 4);  // reused per attention
  u16*   Pm = (u16*)alloc((size_t)Bb * Lt * Lt * 2);    // reused per attention

  auto gemm = [&](const u16* A, long ldA, long bsA,
                  const u16* Bm, long ldB, long bsB,
                  void* C, long ldC, long bsC,
                  int M, int N, int K, int flags, int batch) {
    dim3 grid((unsigned)(N / BN), (unsigned)(M / BM), (unsigned)batch);
    gemm_nt_bf16<<<grid, 128, 0, stream>>>(A, ldA, bsA, Bm, ldB, bsB, C, ldC, bsC, M, N, K, flags);
  };

  // 1) convert activations / embeddings, transpose+convert weights
  { long n = Mt * Dd; cvt_f32_bf16<<<dim3((unsigned)((n + 255) / 256)), 256, 0, stream>>>(text, textb, n); }
  { long n = Mh * Dd; cvt_f32_bf16<<<dim3((unsigned)((n + 255) / 256)), 256, 0, stream>>>(html, htmlb, n); }
  { long n = (long)Ee * Dd; cvt_f32_bf16<<<dim3((unsigned)((n + 255) / 256)), 256, 0, stream>>>(edge, edgeb, n); }
  for (int i = 0; i < 10; ++i)
    transpose_f32_bf16<<<dim3((Dd * Dd + 255) / 256), 256, 0, stream>>>(
        W[i], Wt + (size_t)i * Dd * Dd, Dd, Dd);

  // 2) projections (bf16 output): Wt holds W^T so NT-gemm computes X @ W
  const int tw[5] = {0, 5, 6, 8, 9};  // V_T, K_H2T, Q_T2H, Q_T2T, K_T2T
  const int hw[5] = {1, 2, 3, 4, 7};  // V_H, Q_H2H, K_H2H, Q_H2T, K_T2H
  for (int i = 0; i < 5; ++i)
    gemm(textb, Dd, 0, Wt + (size_t)tw[i] * Dd * Dd, Dd, 0, Pt[i], Dd, 0, (int)Mt, Dd, Dd, 2, 1);
  for (int i = 0; i < 5; ++i)
    gemm(htmlb, Dd, 0, Wt + (size_t)hw[i] * Dd * Dd, Dd, 0, Ph[i], Dd, 0, (int)Mh, Dd, Dd, 2, 1);

  // 3) V transposes (per batch) for use as B-operand in P@V
  transpose_bf16<<<dim3((Lt * Dd + 255) / 256, 1, Bb), 256, 0, stream>>>(
      Pt[0], VtT, Lt, Dd, (long)Lt * Dd, (long)Dd * Lt);
  transpose_bf16<<<dim3((Lh * Dd + 255) / 256, 1, Bb), 256, 0, stream>>>(
      Ph[0], VhT, Lh, Dd, (long)Lh * Dd, (long)Dd * Lh);

  // 4) qe = h2h_q @ edge^T  (f32), qp = t2t_q @ pos^T (f32, N=13 via VALU)
  gemm(Ph[1], Dd, (long)Lh * Dd, edgeb, Dd, 0, qe, Ee, (long)Lh * Ee, Lh, Ee, Dd, 0, Bb);
  { long n = Mt * Pw; qp_kernel<<<dim3((unsigned)((n + 255) / 256)), 256, 0, stream>>>(Pt[3], pose, qp, n); }

  const long obs = (long)(Lt + Lh) * Dd;   // d_out batch stride
  float* outText = out;                    // rows [0, Lt)
  float* outHtml = out + (long)Lt * Dd;    // rows [Lt, Lt+Lh)

  // 5) H2H: scores -> softmax(+edge bias, mask==0) -> P@Vh into html rows
  gemm(Ph[1], Dd, (long)Lh * Dd, Ph[2], Dd, (long)Lh * Dd, S, Lh, (long)Lh * Lh, Lh, Lh, Dd, 0, Bb);
  softmax_kernel<<<dim3(Lh, Bb), 256, 0, stream>>>(S, Pm, H2H, nullptr, qe, Ee, Lh, Lh, 2);
  gemm(Pm, Lh, (long)Lh * Lh, VhT, Lh, (long)Dd * Lh, outHtml, Dd, obs, Lh, Dd, Lh, 0, Bb);

  // 6) H2T: scores -> softmax(float mask) -> P@Vt accumulated into html rows
  gemm(Ph[3], Dd, (long)Lh * Dd, Pt[1], Dd, (long)Lt * Dd, S, Lt, (long)Lh * Lt, Lh, Lt, Dd, 0, Bb);
  softmax_kernel<<<dim3(Lh, Bb), 256, 0, stream>>>(S, Pm, nullptr, H2T, nullptr, 0, Lh, Lt, 1);
  gemm(Pm, Lt, (long)Lh * Lt, VtT, Lt, (long)Dd * Lt, outHtml, Dd, obs, Lh, Dd, Lt, 1, Bb);

  // 7) T2H: scores -> plain softmax -> P@Vh into text rows
  gemm(Pt[2], Dd, (long)Lt * Dd, Ph[4], Dd, (long)Lh * Dd, S, Lh, (long)Lt * Lh, Lt, Lh, Dd, 0, Bb);
  softmax_kernel<<<dim3(Lt, Bb), 256, 0, stream>>>(S, Pm, nullptr, nullptr, nullptr, 0, Lt, Lh, 0);
  gemm(Pm, Lh, (long)Lt * Lh, VhT, Lh, (long)Dd * Lh, outText, Dd, obs, Lt, Dd, Lh, 0, Bb);

  // 8) T2T: scores -> softmax(+pos bias, float mask) -> P@Vt accumulated into text rows
  gemm(Pt[3], Dd, (long)Lt * Dd, Pt[4], Dd, (long)Lt * Dd, S, Lt, (long)Lt * Lt, Lt, Lt, Dd, 0, Bb);
  softmax_kernel<<<dim3(Lt, Bb), 256, 0, stream>>>(S, Pm, nullptr, T2T, qp, Pw, Lt, Lt, 3);
  gemm(Pm, Lt, (long)Lt * Lt, VtT, Lt, (long)Dd * Lt, outText, Dd, obs, Lt, Dd, Lt, 1, Bb);
}
